// PowerAttentionLayer_54735063220361
// MI455X (gfx1250) — compile-verified, matrix operand order checked by reading, not run on hardware
//
#include <hip/hip_runtime.h>
#include <hip/hip_bf16.h>
#include <math.h>

// ---------------------------------------------------------------------------
// Power attention on gfx1250 (MI455X): chunked linear-attention reformulation
// so every heavy op is a v_wmma_f32_16x16x32_bf16.
// ---------------------------------------------------------------------------

#define B_DIM   2
#define S_DIM   1024
#define HID     512
#define NH      16
#define HD      32
#define SP      528        // 32 squares + 496 cross terms
#define SPAD    544        // padded to 17 * 32 for WMMA K-steps
#define CH      32         // chunk length (time)
#define NCHUNK  (S_DIM / CH)
#define GAMMA_F 0.9f
#define SQRT2F  1.41421356237309504880f

// GEMM dims are fixed for every projection in this model.
#define GM      2048
#define GN      512
#define GK      512
#define BM      128        // block tile M (8 waves x 32 rows / pair layout)
#define BN      64         // block tile N
#define BK      32         // one WMMA K-step

typedef __attribute__((ext_vector_type(16))) __bf16 v16bf;
typedef __attribute__((ext_vector_type(8)))  __bf16 v8bf;
typedef __attribute__((ext_vector_type(8)))  float  v8f;

__device__ __forceinline__ v8f wmma_bf16(v16bf a, v16bf b, v8f c) {
  // D = A(16x32 bf16) * B(32x16 bf16) + C(16x16 f32)
  return __builtin_amdgcn_wmma_f32_16x16x32_bf16(
      /*neg_a=*/false, a, /*neg_b=*/false, b,
      /*c_mod=*/(short)0, c, /*reuse_a=*/false, /*reuse_b=*/false);
}

// 16x32 bf16 fragment from LDS, row-major [m][k] (A-operand), also used for
// B-operands whose backing store is laid out [n][k] row-major.
// ISA layout: lanes 0-15 -> K 0..7 / 16..23, lanes 16-31 -> K 8..15 / 24..31.
__device__ __forceinline__ v16bf load_frag_bf(const __bf16* base, int row0,
                                              int stride, int k0, int lane) {
  int m    = lane & 15;
  int koff = (lane >> 4) << 3;
  const __bf16* p = base + (row0 + m) * stride + k0 + koff;
  v8bf lo = *(const v8bf*)p;         // K = k0+koff .. +7   (ds_load_b128)
  v8bf hi = *(const v8bf*)(p + 16);  // K = k0+koff+16 .. +23
  v16bf r;
#pragma unroll
  for (int e = 0; e < 8; ++e) { r[e] = lo[e]; r[e + 8] = hi[e]; }
  return r;
}

// Same fragment but converting from an fp32 backing store (the recurrent state
// is kept fp32 in LDS for accuracy; cast to bf16 only at the WMMA boundary).
__device__ __forceinline__ v16bf load_frag_f32cvt(const float* base, int row0,
                                                  int stride, int k0, int lane) {
  int m    = lane & 15;
  int koff = (lane >> 4) << 3;
  const float* p = base + (row0 + m) * stride + k0 + koff;
  float4 a0 = *(const float4*)p;
  float4 a1 = *(const float4*)(p + 4);
  float4 b0 = *(const float4*)(p + 16);
  float4 b1 = *(const float4*)(p + 20);
  v16bf r;
  r[0]  = (__bf16)a0.x; r[1]  = (__bf16)a0.y; r[2]  = (__bf16)a0.z; r[3]  = (__bf16)a0.w;
  r[4]  = (__bf16)a1.x; r[5]  = (__bf16)a1.y; r[6]  = (__bf16)a1.z; r[7]  = (__bf16)a1.w;
  r[8]  = (__bf16)b0.x; r[9]  = (__bf16)b0.y; r[10] = (__bf16)b0.z; r[11] = (__bf16)b0.w;
  r[12] = (__bf16)b1.x; r[13] = (__bf16)b1.y; r[14] = (__bf16)b1.z; r[15] = (__bf16)b1.w;
  return r;
}

// Map linear upper-triangle index c (np.triu_indices(32, k=1) order) -> (i, j).
// Only used once at kernel start to fill the LDS pair table.
__device__ __forceinline__ void triu_pair(int c, int& ai, int& bj) {
  int a = (int)((63.0f - sqrtf(3969.0f - 8.0f * (float)c)) * 0.5f);
  if (a < 0) a = 0; if (a > 30) a = 30;
  int base = (a * (63 - a)) >> 1;
  while (base > c)           { --a; base = (a * (63 - a)) >> 1; }
  while (c - base >= 31 - a) { base += 31 - a; ++a; }
  ai = a;
  bj = a + 1 + (c - base);
}

// ---------------------------------------------------------------------------
// Kernel 1/3: Y = X @ W + bias (2048x512 @ 512x512), bf16 WMMA, fp32 accum.
// BM=128, BN=64, BK=32; each of 8 waves owns a 32x32 output tile (2x2 WMMA
// tiles -> 4 WMMAs from 4 fragment loads, each fragment reused twice).
// Double-buffered LDS; W staged transposed so B-fragments are contiguous-K.
// ---------------------------------------------------------------------------
__global__ __launch_bounds__(256) void gemm_bias_kernel(
    const float* __restrict__ X, const float* __restrict__ W,
    const float* __restrict__ bias, float* __restrict__ Y) {
  __shared__ __bf16 lA[2][BM * BK];   // [m][k]
  __shared__ __bf16 lB[2][BN * BK];   // [n][k] (transposed on the way in)

  const int tid  = threadIdx.x;
  const int lane = tid & 31;
  const int w    = tid >> 5;
  const int m0   = blockIdx.y * BM;
  const int n0   = blockIdx.x * BN;
  const int wr0  = (w & 3) * 32;       // wave's 32-row strip within block
  const int wc0  = (w >> 2) * 32;      // wave's 32-col strip within block

  // A stage: 4096 elems / 256 thr = 16 each (one half-row of 16 floats)
  const int arow = tid >> 1, ac0 = (tid & 1) << 4;
  // B stage: 2048 elems / 256 thr = 8 each (8 n-values for one k), transposed
  const int bkk  = tid >> 3, bn0 = (tid & 7) << 3;

  const float* gpA = X + (size_t)(m0 + arow) * GK + ac0;
  const float* gpB = W + (size_t)bkk * GN + n0 + bn0;

  auto stage = [&](int b, float4 a0, float4 a1, float4 a2, float4 a3,
                   float4 b0v, float4 b1v) {
    __bf16* dp = lA[b] + arow * BK + ac0;
    dp[0]  = (__bf16)a0.x; dp[1]  = (__bf16)a0.y; dp[2]  = (__bf16)a0.z; dp[3]  = (__bf16)a0.w;
    dp[4]  = (__bf16)a1.x; dp[5]  = (__bf16)a1.y; dp[6]  = (__bf16)a1.z; dp[7]  = (__bf16)a1.w;
    dp[8]  = (__bf16)a2.x; dp[9]  = (__bf16)a2.y; dp[10] = (__bf16)a2.z; dp[11] = (__bf16)a2.w;
    dp[12] = (__bf16)a3.x; dp[13] = (__bf16)a3.y; dp[14] = (__bf16)a3.z; dp[15] = (__bf16)a3.w;
    __bf16* bp = lB[b] + bkk;
    bp[(bn0 + 0) * BK] = (__bf16)b0v.x;
    bp[(bn0 + 1) * BK] = (__bf16)b0v.y;
    bp[(bn0 + 2) * BK] = (__bf16)b0v.z;
    bp[(bn0 + 3) * BK] = (__bf16)b0v.w;
    bp[(bn0 + 4) * BK] = (__bf16)b1v.x;
    bp[(bn0 + 5) * BK] = (__bf16)b1v.y;
    bp[(bn0 + 6) * BK] = (__bf16)b1v.z;
    bp[(bn0 + 7) * BK] = (__bf16)b1v.w;
  };

  float4 fa0 = *(const float4*)gpA;
  float4 fa1 = *(const float4*)(gpA + 4);
  float4 fa2 = *(const float4*)(gpA + 8);
  float4 fa3 = *(const float4*)(gpA + 12);
  float4 fb0 = *(const float4*)gpB;
  float4 fb1 = *(const float4*)(gpB + 4);

  v8f acc00 = {}, acc01 = {}, acc10 = {}, acc11 = {};
  stage(0, fa0, fa1, fa2, fa3, fb0, fb1);
  __syncthreads();

  int buf = 0;
#pragma unroll 1
  for (int kb = 0; kb < GK / BK; ++kb) {
    if (kb + 1 < GK / BK) {            // register-prefetch next K tile
      const float* pA = gpA + (kb + 1) * BK;
      const float* pB = gpB + (size_t)(kb + 1) * BK * GN;
      fa0 = *(const float4*)pA;       fa1 = *(const float4*)(pA + 4);
      fa2 = *(const float4*)(pA + 8); fa3 = *(const float4*)(pA + 12);
      fb0 = *(const float4*)pB;       fb1 = *(const float4*)(pB + 4);
    }
    v16bf a0 = load_frag_bf(lA[buf], wr0 + 0,  BK, 0, lane);
    v16bf a1 = load_frag_bf(lA[buf], wr0 + 16, BK, 0, lane);
    v16bf b0 = load_frag_bf(lB[buf], wc0 + 0,  BK, 0, lane);
    v16bf b1 = load_frag_bf(lB[buf], wc0 + 16, BK, 0, lane);
    acc00 = wmma_bf16(a0, b0, acc00);
    acc01 = wmma_bf16(a0, b1, acc01);
    acc10 = wmma_bf16(a1, b0, acc10);
    acc11 = wmma_bf16(a1, b1, acc11);
    if (kb + 1 < GK / BK) {
      buf ^= 1;
      stage(buf, fa0, fa1, fa2, fa3, fb0, fb1);  // other buffer: no pre-sync
      __syncthreads();                           // one barrier per K-step
    }
  }

  const int mloc = (lane >> 4) << 3;
  const int nloc = lane & 15;
  const int gm   = m0 + wr0 + mloc;
  const int gn0  = n0 + wc0 + nloc;
  const int gn1  = gn0 + 16;
  const float b0s = bias[gn0];
  const float b1s = bias[gn1];
  float* yp = Y + (size_t)gm * GN;
#pragma unroll
  for (int r = 0; r < 8; ++r) {        // row stride folds into IOFFSET imm
    yp[r * GN + gn0]          = acc00[r] + b0s;
    yp[r * GN + gn1]          = acc01[r] + b1s;
    yp[(r + 16) * GN + gn0]   = acc10[r] + b0s;
    yp[(r + 16) * GN + gn1]   = acc11[r] + b1s;
  }
}

// ---------------------------------------------------------------------------
// Kernel 2/3: chunked power-attention recurrence. One WG per (b, h); 8 waves.
// ---------------------------------------------------------------------------
#define RECUR_LDS_BYTES (32*SPAD*4 /*stateT*/ + 32*32*4 /*oacc*/ + \
                         32*32*4*2 /*qtmp,ktmp*/ + 160 /*gpow*/ + \
                         32*SPAD*2*3 /*sq,sk,skT*/ + 32*32*2*3 /*vT,vsT,aM*/ + \
                         1024 /*pair table*/)

__global__ __launch_bounds__(256) void recurrence_kernel(
    const float* __restrict__ Q, const float* __restrict__ Km,
    const float* __restrict__ V, float* __restrict__ Out) {
  extern __shared__ char smem_raw[];
  float*  stateT = (float*)smem_raw;          // [32 n][544 d] fp32 state^T
  float*  oacc   = stateT + 32 * SPAD;        // [32 i][32 n]
  float*  qtmp   = oacc + 32 * 32;            // [32 i][32 d]
  float*  ktmp   = qtmp + 32 * 32;            // [32 i][32 d]
  float*  gpow   = ktmp + 32 * 32;            // gamma^t, t = 0..32 (40 slots)
  __bf16* sq     = (__bf16*)(gpow + 40);      // [32 i][544 d]
  __bf16* sk     = sq  + 32 * SPAD;           // [32 j][544 d]
  __bf16* skT    = sk  + 32 * SPAD;           // [544 d][32 j]
  __bf16* vT     = skT + SPAD * 32;           // [32 n][32 j]
  __bf16* vsT    = vT  + 32 * 32;             // [32 n][32 j], gamma-scaled
  __bf16* aM     = vsT + 32 * 32;             // [32 i][32 j] masked attn
  unsigned short* pairs = (unsigned short*)(aM + 32 * 32);  // 496 (i,j) pairs

  const int tid  = threadIdx.x;
  const int lane = tid & 31;
  const int w    = tid >> 5;
  const int bb   = blockIdx.x / NH;
  const int hh   = blockIdx.x % NH;
  const int mloc = (lane >> 4) << 3;
  const int nloc = lane & 15;

  // One-time tables: recurrent state = 0, gamma powers, triu pair map.
  for (int idx = tid; idx < 32 * SPAD; idx += 256) stateT[idx] = 0.0f;
  if (tid < 33) gpow[tid] = __powf(GAMMA_F, (float)tid);
  for (int c = tid; c < 496; c += 256) {
    int ai, bj; triu_pair(c, ai, bj);
    pairs[c] = (unsigned short)(ai | (bj << 8));
  }
  __syncthreads();
  const float gT = gpow[CH];

  const int si  = tid >> 3;          // stage: time row handled by this thread
  const int sd0 = (tid & 7) * 4;     // stage: 4 dims
  const int pi  = tid >> 3;          // spow: time row
  const int pd0 = (tid & 7) * 68;    // spow: 68 feature slots

  for (int cc = 0; cc < NCHUNK; ++cc) {
    const int t0 = cc * CH;

    // ---- stage q/k chunk to LDS; build vT / gamma-scaled vsT directly -----
    {
      size_t base = ((size_t)(bb * S_DIM + t0 + si)) * HID + hh * HD + sd0;
      *(float4*)(qtmp + si * 32 + sd0) = *(const float4*)(Q + base);
      *(float4*)(ktmp + si * 32 + sd0) = *(const float4*)(Km + base);
      float4 vv = *(const float4*)(V + base);
      float sc = gpow[CH - 1 - si];
      vT[(sd0 + 0) * 32 + si] = (__bf16)vv.x; vsT[(sd0 + 0) * 32 + si] = (__bf16)(vv.x * sc);
      vT[(sd0 + 1) * 32 + si] = (__bf16)vv.y; vsT[(sd0 + 1) * 32 + si] = (__bf16)(vv.y * sc);
      vT[(sd0 + 2) * 32 + si] = (__bf16)vv.z; vsT[(sd0 + 2) * 32 + si] = (__bf16)(vv.z * sc);
      vT[(sd0 + 3) * 32 + si] = (__bf16)vv.w; vsT[(sd0 + 3) * 32 + si] = (__bf16)(vv.w * sc);
      if (cc + 1 < NCHUNK) {
        size_t nb = base + (size_t)CH * HID;
        __builtin_prefetch(Q + nb, 0, 1);   // global_prefetch_b8
        __builtin_prefetch(Km + nb, 0, 1);
        __builtin_prefetch(V + nb, 0, 1);
      }
    }
    __syncthreads();

    // ---- spow2 feature expansion (d=32 -> 544 padded), sk also transposed --
    {
      const float* qi = qtmp + pi * 32;
      const float* ki = ktmp + pi * 32;
      for (int jj = 0; jj < 68; ++jj) {
        int dd = pd0 + jj;
        float sv, kv;
        if (dd < HD)      { float a = qi[dd]; sv = a * a; float c = ki[dd]; kv = c * c; }
        else if (dd < SP) { int pc = pairs[dd - HD];
                            int ai = pc & 255, bj = pc >> 8;
                            sv = SQRT2F * qi[ai] * qi[bj];
                            kv = SQRT2F * ki[ai] * ki[bj]; }
        else              { sv = 0.0f; kv = 0.0f; }
        sq[pi * SPAD + dd] = (__bf16)sv;
        __bf16 kb16 = (__bf16)kv;
        sk[pi * SPAD + dd] = kb16;
        skT[dd * 32 + pi]  = kb16;
      }
    }
    __syncthreads();

    // ---- phase 1: waves 0-3 SQ @ state^T ; waves 4-7 A = SQ @ SK^T --------
    if (w < 4) {
      int tr = w >> 1, tc = w & 1;
      v8f acc = {};
      for (int k0 = 0; k0 < SPAD; k0 += 32) {
        v16bf a = load_frag_bf(sq, tr * 16, SPAD, k0, lane);
        v16bf b = load_frag_f32cvt(stateT, tc * 16, SPAD, k0, lane);
        acc = wmma_bf16(a, b, acc);
      }
#pragma unroll
      for (int r = 0; r < 8; ++r) {
        int i = tr * 16 + mloc + r;
        oacc[i * 32 + tc * 16 + nloc] = acc[r] * gpow[i + 1];
      }
    } else {
      int w2 = w - 4, tr = w2 >> 1, tc = w2 & 1;
      v8f acc = {};
      for (int k0 = 0; k0 < SPAD; k0 += 32) {
        v16bf a = load_frag_bf(sq, tr * 16, SPAD, k0, lane);
        v16bf b = load_frag_bf(sk, tc * 16, SPAD, k0, lane);
        acc = wmma_bf16(a, b, acc);
      }
#pragma unroll
      for (int r = 0; r < 8; ++r) {
        int i = tr * 16 + mloc + r;
        int j = tc * 16 + nloc;
        float v = (j <= i) ? acc[r] * gpow[i - j] : 0.0f;
        aM[i * 32 + j] = (__bf16)v;
      }
    }
    __syncthreads();

    // ---- phase 2: waves 0-3 out = oacc + A @ V ; all waves update state ---
    if (w < 4) {
      int tr = w >> 1, tc = w & 1;
      v16bf a = load_frag_bf(aM, tr * 16, 32, 0, lane);
      v16bf b = load_frag_bf(vT, tc * 16, 32, 0, lane);
      v8f acc = {};
      acc = wmma_bf16(a, b, acc);
#pragma unroll
      for (int r = 0; r < 8; ++r) {
        int i = tr * 16 + mloc + r;
        int n = tc * 16 + nloc;
        Out[((size_t)(bb * S_DIM + t0 + i)) * HID + hh * HD + n] =
            oacc[i * 32 + n] + acc[r];
      }
    }
    // state^T[n][d] = gamma^T * state^T + (scaled V)^T @ SK : 68 tiles / 8 waves
    for (int t = w; t < 68; t += 8) {
      int d0    = (t >> 1) * 16;
      int nrow0 = (t & 1) * 16;
      v16bf a = load_frag_bf(vsT, nrow0, 32, 0, lane);
      v16bf b = load_frag_bf(skT, d0, 32, 0, lane);
      v8f acc = {};
      acc = wmma_bf16(a, b, acc);
#pragma unroll
      for (int r = 0; r < 8; ++r) {
        int n = nrow0 + mloc + r;
        int d = d0 + nloc;
        int idx = n * SPAD + d;
        stateT[idx] = gT * stateT[idx] + acc[r];
      }
    }
    __syncthreads();
  }
}

// ---------------------------------------------------------------------------
extern "C" void kernel_launch(void* const* d_in, const int* in_sizes, int n_in,
                              void* d_out, int out_size, void* d_ws, size_t ws_size,
                              hipStream_t stream) {
  const float* x  = (const float*)d_in[0];
  const float* qw = (const float*)d_in[1];
  const float* qb = (const float*)d_in[2];
  const float* kw = (const float*)d_in[3];
  const float* kb = (const float*)d_in[4];
  const float* vw = (const float*)d_in[5];
  const float* vb = (const float*)d_in[6];
  const float* ow = (const float*)d_in[7];
  const float* ob = (const float*)d_in[8];
  float* out = (float*)d_out;

  const size_t MK = (size_t)B_DIM * S_DIM * HID;  // 2048*512 = 1M floats
  float* qbuf = (float*)d_ws;
  float* kbuf = qbuf + MK;
  float* vbuf = kbuf + MK;
  float* abuf = vbuf + MK;   // attention output before final projection

  dim3 gg(GN / BN, GM / BM);

  gemm_bias_kernel<<<gg, 256, 0, stream>>>(x, qw, qb, qbuf);
  gemm_bias_kernel<<<gg, 256, 0, stream>>>(x, kw, kb, kbuf);
  gemm_bias_kernel<<<gg, 256, 0, stream>>>(x, vw, vb, vbuf);

  (void)hipFuncSetAttribute((const void*)recurrence_kernel,
                            hipFuncAttributeMaxDynamicSharedMemorySize,
                            RECUR_LDS_BYTES);
  recurrence_kernel<<<B_DIM * NH, 256, RECUR_LDS_BYTES, stream>>>(
      qbuf, kbuf, vbuf, abuf);

  gemm_bias_kernel<<<gg, 256, 0, stream>>>(abuf, ow, ob, out);
}
